// MultiDecoder_42468636623052
// MI455X (gfx1250) — compile-verified, matrix operand order checked by reading, not run on hardware
//
#include <hip/hip_runtime.h>
#include <math.h>

// ---------------- problem constants ----------------
#define BB 256
#define SS 100
#define HH 512
#define EE 512
#define RR 50
#define NCELL 5

typedef __attribute__((ext_vector_type(16))) __bf16 bf16x16;
typedef __attribute__((ext_vector_type(8)))  float  f32x8;

// f32 -> bf16 (round to nearest even)
__device__ __forceinline__ unsigned short f2bf(float f) {
  unsigned int u = __float_as_uint(f);
  unsigned int r = (u + 0x7FFFu + ((u >> 16) & 1u)) >> 16;
  return (unsigned short)r;
}
__device__ __forceinline__ float seluf(float x) {
  const float sc = 1.0507009873554805f, al = 1.6732632423543772f;
  return x > 0.f ? sc * x : sc * al * (expf(x) - 1.f);
}
__device__ __forceinline__ float sigm(float x) { return 1.f / (1.f + expf(-x)); }

// =====================================================================
// Generic bf16 WMMA GEMM: C[M,N] = A[M,K] * B[N,K]^T (+bias[n]) (+Cadd[M,N])
// K multiple of 32, rows 16B aligned (K=512 at every call site).
// Wave tile: 16(M) x 64(N) = 4 accumulators, A-fragment reused.
// =====================================================================
__global__ __launch_bounds__(256) void k_gemm_bf16(
    const __bf16* __restrict__ A, const __bf16* __restrict__ B,
    float* __restrict__ C, unsigned short* __restrict__ Cbf,
    const float* __restrict__ bias, const float* __restrict__ Cadd,
    int M, int N, int K) {
  int ntiles = (N + 63) >> 6;
  int mtiles = M >> 4;
  int wave = blockIdx.x * (blockDim.x >> 5) + (threadIdx.x >> 5);
  if (wave >= ntiles * mtiles) return;
  int mt = wave / ntiles, nt = wave % ntiles;
  int lane = threadIdx.x & 31;
  int half = lane >> 4;
  int l16 = lane & 15;

  const __bf16* pA = A + (size_t)(mt * 16 + l16) * K;
  const __bf16* pB[4];
  for (int s = 0; s < 4; ++s) {
    int n = nt * 64 + s * 16 + l16;
    if (n > N - 1) n = N - 1;
    pB[s] = B + (size_t)n * K;
  }

  f32x8 acc[4];
  for (int s = 0; s < 4; ++s)
    for (int r = 0; r < 8; ++r) acc[s][r] = 0.f;

  for (int k0 = 0; k0 < K; k0 += 32) {
    int ka = k0 + half * 8;
    int kb = ka + 16;
    union { bf16x16 v; uint4 q[2]; } af;
    af.q[0] = *(const uint4*)(pA + ka);
    af.q[1] = *(const uint4*)(pA + kb);
#pragma unroll
    for (int s = 0; s < 4; ++s) {
      union { bf16x16 v; uint4 q[2]; } bf;
      bf.q[0] = *(const uint4*)(pB[s] + ka);
      bf.q[1] = *(const uint4*)(pB[s] + kb);
      acc[s] = __builtin_amdgcn_wmma_f32_16x16x32_bf16(
          false, af.v, false, bf.v, (short)0, acc[s], false, false);
    }
  }

  for (int s = 0; s < 4; ++s) {
    int n = nt * 64 + s * 16 + l16;
    if (n >= N) continue;
    float bv = bias ? bias[n] : 0.f;
#pragma unroll
    for (int r = 0; r < 8; ++r) {
      int m = mt * 16 + r + 8 * half;
      float v = acc[s][r] + bv;
      size_t idx = (size_t)m * N + n;
      if (Cadd) v += Cadd[idx];
      if (C) C[idx] = v;
      if (Cbf) Cbf[idx] = f2bf(v);
    }
  }
}

// ---------------- small utility kernels ----------------
__global__ void k_conv_bf16(const float* __restrict__ src, unsigned short* __restrict__ dst,
                            int R, int C, int ld, int c0) {
  int i = blockIdx.x * blockDim.x + threadIdx.x;
  if (i >= R * C) return;
  int r = i / C, c = i - r * C;
  dst[i] = f2bf(src[(size_t)r * ld + c + c0]);
}
__global__ void k_selu_bf16(const float* __restrict__ src, unsigned short* __restrict__ dst, int n) {
  int i = blockIdx.x * blockDim.x + threadIdx.x;
  if (i < n) dst[i] = f2bf(seluf(src[i]));
}
__global__ void k_bpe(const float* bp, const float* be, float* bpe) {
  int i = threadIdx.x;
  if (i < RR) bpe[i] = bp[i];
  else if (i == RR) bpe[i] = be[0];
}
__global__ void k_zero(float* p, int n) {
  int i = blockIdx.x * blockDim.x + threadIdx.x;
  if (i < n) p[i] = 0.f;
}
__global__ void k_sos(const float* sos, unsigned short* embbf) {
  int i = blockIdx.x * blockDim.x + threadIdx.x;
  if (i < BB * EE) embbf[i] = f2bf(sos[i & (EE - 1)]);
}
__global__ void k_cell_init(const float* dec, float* h, unsigned short* hbf) {
  int i = blockIdx.x * blockDim.x + threadIdx.x;
  if (i >= BB * HH) return;
  float v = 0.5f * (dec[i] + h[i]);
  h[i] = v;
  hbf[i] = f2bf(v);
}

// score[b,s] = dot(enc[b,s,:], W_attn[0, H:2H])   (bias & h-term cancel in softmax)
__global__ __launch_bounds__(256) void k_score(const float* __restrict__ enc,
                                               const float* __restrict__ Wattn,
                                               float* __restrict__ score) {
  int wid = blockIdx.x * (blockDim.x >> 5) + (threadIdx.x >> 5);
  int lane = threadIdx.x & 31;
  if (wid >= BB * SS) return;
  const float* e = enc + (size_t)wid * HH;
  float s = 0.f;
  for (int h = lane; h < HH; h += 32) s += e[h] * Wattn[HH + h];
  for (int o = 16; o; o >>= 1) s += __shfl_xor(s, o);
  if (lane == 0) score[wid] = s;
}

// softmax over S + context[b,:] -> bf16
__global__ __launch_bounds__(256) void k_ctx(const float* __restrict__ score,
                                             const float* __restrict__ enc,
                                             unsigned short* __restrict__ ctxbf) {
  __shared__ float aw[SS];
  int b = blockIdx.x, tid = threadIdx.x;
  if (tid == 0) {
    float mx = -1e30f;
    for (int s = 0; s < SS; ++s) mx = fmaxf(mx, score[b * SS + s]);
    float sum = 0.f;
    for (int s = 0; s < SS; ++s) { float e = expf(score[b * SS + s] - mx); aw[s] = e; sum += e; }
    float inv = 1.f / sum;
    for (int s = 0; s < SS; ++s) aw[s] *= inv;
  }
  __syncthreads();
  for (int h = tid; h < HH; h += blockDim.x) {
    float c = 0.f;
    for (int s = 0; s < SS; ++s) c += aw[s] * enc[((size_t)b * SS + s) * HH + h];
    ctxbf[b * HH + h] = f2bf(c);
  }
}

// GRU elementwise: h = (1-z)*n + z*h ; also write bf16(h), bf16(selu(h))
__global__ void k_gru(const float* __restrict__ gi, const float* __restrict__ gh,
                      float* __restrict__ h, unsigned short* __restrict__ hbf,
                      unsigned short* __restrict__ shbf) {
  int i = blockIdx.x * blockDim.x + threadIdx.x;
  if (i >= BB * HH) return;
  int b = i / HH, j = i - b * HH;
  const float* gib = gi + (size_t)b * 3 * HH;
  const float* ghb = gh + (size_t)b * 3 * HH;
  float r = sigm(gib[j] + ghb[j]);
  float z = sigm(gib[HH + j] + ghb[HH + j]);
  float n = tanhf(gib[2 * HH + j] + r * ghb[2 * HH + j]);
  float hn = (1.f - z) * n + z * h[i];
  h[i] = hn;
  hbf[i] = f2bf(hn);
  shbf[i] = f2bf(seluf(hn));
}

// eos[b] = h[b,:]·W_eos + b_eos   (wave per row)
__global__ __launch_bounds__(256) void k_eos(const float* __restrict__ h,
                                             const float* __restrict__ Weos,
                                             const float* __restrict__ beos,
                                             float* __restrict__ eos) {
  int wid = blockIdx.x * (blockDim.x >> 5) + (threadIdx.x >> 5);
  int lane = threadIdx.x & 31;
  if (wid >= BB) return;
  const float* p = h + (size_t)wid * HH;
  float s = 0.f;
  for (int j = lane; j < HH; j += 32) s += p[j] * Weos[j];
  for (int o = 16; o; o >>= 1) s += __shfl_xor(s, o);
  if (lane == 0) eos[wid] = s + beos[0];
}

// pred finalize: log_softmax over 51, first-max argmax, rel_emb gather index
__global__ __launch_bounds__(256) void k_pred_fin(const float* __restrict__ pred,
                                                  float* __restrict__ out_rel,
                                                  float* __restrict__ out_act,
                                                  int* __restrict__ act) {
  int b = blockIdx.x * (blockDim.x >> 5) + (threadIdx.x >> 5);
  int lane = threadIdx.x & 31;
  if (b >= BB) return;
  const float* p = pred + (size_t)b * (RR + 1);
  float v0 = (lane < RR + 1) ? p[lane] : -1e30f;
  float v1 = (lane + 32 < RR + 1) ? p[lane + 32] : -1e30f;
  float mv = fmaxf(v0, v1);
  int mi = (v1 > v0) ? lane + 32 : lane;
  float mx = mv;
  for (int o = 16; o; o >>= 1) mx = fmaxf(mx, __shfl_xor(mx, o));
  float se = ((lane < RR + 1) ? expf(v0 - mx) : 0.f) + ((lane + 32 < RR + 1) ? expf(v1 - mx) : 0.f);
  for (int o = 16; o; o >>= 1) se += __shfl_xor(se, o);
  float lse = mx + logf(se);
  if (lane < RR + 1) out_rel[(size_t)b * (RR + 1) + lane] = v0 - lse;
  if (lane + 32 < RR + 1) out_rel[(size_t)b * (RR + 1) + lane + 32] = v1 - lse;
  for (int o = 16; o; o >>= 1) {
    float ov = __shfl_xor(mv, o); int oi = __shfl_xor(mi, o);
    if (ov > mv || (ov == mv && oi < mi)) { mv = ov; mi = oi; }
  }
  if (lane == 0) { out_act[b] = (float)mi; act[b] = mi; }
}

// copy finalize: selu-dot epilogue, concat eos, log_softmax(101), argmax, word id
__global__ __launch_bounds__(128) void k_copy_fin(
    const float* __restrict__ outF, const float* __restrict__ encF,
    const float* __restrict__ Wcopy, const float* __restrict__ bcopy,
    const float* __restrict__ eos, const int* __restrict__ sentence,
    float* __restrict__ out_copy, float* __restrict__ out_act, int* __restrict__ word) {
  __shared__ float so[SS], sw[SS], sv[SS + 1];
  __shared__ float red[128];
  __shared__ int redi[128];
  int b = blockIdx.x, tid = threadIdx.x;
  if (tid < SS) { so[tid] = outF[b * SS + tid]; sw[tid] = Wcopy[tid]; }
  __syncthreads();
  if (tid < SS) {
    const float* er = encF + ((size_t)b * SS + tid) * SS;
    float a = 0.f;
    for (int k = 0; k < SS; ++k) a += seluf(so[k] + er[k]) * sw[k];
    sv[tid] = a + bcopy[0];
  }
  if (tid == SS) sv[SS] = eos[b];
  __syncthreads();
  float v = (tid <= SS) ? sv[tid] : -1e30f;
  red[tid] = v; redi[tid] = (tid <= SS) ? tid : (SS + 2);
  __syncthreads();
  for (int o = 64; o; o >>= 1) {
    if (tid < o) {
      if (red[tid + o] > red[tid] || (red[tid + o] == red[tid] && redi[tid + o] < redi[tid])) {
        red[tid] = red[tid + o]; redi[tid] = redi[tid + o];
      }
    }
    __syncthreads();
  }
  float mx = red[0]; int amax = redi[0];
  __syncthreads();
  red[tid] = (tid <= SS) ? expf(sv[tid] - mx) : 0.f;
  __syncthreads();
  for (int o = 64; o; o >>= 1) {
    if (tid < o) red[tid] += red[tid + o];
    __syncthreads();
  }
  float lse = mx + logf(red[0]);
  if (tid <= SS) out_copy[(size_t)b * (SS + 1) + tid] = sv[tid] - lse;
  if (tid == 0) {
    out_act[b] = (float)amax;
    int s = amax; if (s > SS - 1) s = SS - 1; if (s < 0) s = 0;
    word[b] = sentence[b * SS + s];
  }
}

// gather table[idx[b],:] -> bf16 embedding buffer
__global__ __launch_bounds__(128) void k_gather(const float* __restrict__ table,
                                                const int* __restrict__ idx,
                                                unsigned short* __restrict__ embbf) {
  int b = blockIdx.x;
  const float* row = table + (size_t)idx[b] * EE;
  for (int e = threadIdx.x; e < EE; e += blockDim.x)
    embbf[(size_t)b * EE + e] = f2bf(row[e]);
}

// =====================================================================
extern "C" void kernel_launch(void* const* d_in, const int* in_sizes, int n_in,
                              void* d_out, int out_size, void* d_ws, size_t ws_size,
                              hipStream_t stream) {
  const int* sentence    = (const int*)d_in[0];
  const float* dec_state = (const float*)d_in[1];
  const float* enc       = (const float*)d_in[2];
  const float* word_emb  = (const float*)d_in[3];
  const float* rel_emb   = (const float*)d_in[4];
  const float* sos_emb   = (const float*)d_in[5];
  const float* W_comb    = (const float*)d_in[6];
  const float* b_comb    = (const float*)d_in[7];
  const float* W_attn    = (const float*)d_in[8];
  const float* W_ih      = (const float*)d_in[10];
  const float* W_hh      = (const float*)d_in[11];
  const float* b_ih      = (const float*)d_in[12];
  const float* b_hh      = (const float*)d_in[13];
  const float* W_eos     = (const float*)d_in[14];
  const float* b_eos     = (const float*)d_in[15];
  const float* W_pred    = (const float*)d_in[16];
  const float* b_pred    = (const float*)d_in[17];
  const float* W_fuse    = (const float*)d_in[18];
  const float* b_fuse    = (const float*)d_in[19];
  const float* W_copy    = (const float*)d_in[20];
  const float* b_copy    = (const float*)d_in[21];
  float* out = (float*)d_out;

  // bump allocator on d_ws
  char* base = (char*)d_ws;
  size_t off = 0;
  auto alloc = [&](size_t bytes) -> void* {
    void* p = base + off;
    off = (off + bytes + 255) & ~(size_t)255;
    return p;
  };
  unsigned short* WcEmb  = (unsigned short*)alloc((size_t)EE * EE * 2);
  unsigned short* Wcc    = (unsigned short*)alloc((size_t)EE * HH * 2);
  unsigned short* Wih    = (unsigned short*)alloc((size_t)NCELL * 3 * HH * EE * 2);
  unsigned short* Whh    = (unsigned short*)alloc((size_t)NCELL * 3 * HH * HH * 2);
  unsigned short* WfOut  = (unsigned short*)alloc((size_t)100 * HH * 2);
  unsigned short* WfEnc  = (unsigned short*)alloc((size_t)100 * HH * 2);
  unsigned short* Wpe    = (unsigned short*)alloc((size_t)(RR + 1) * HH * 2);
  float* bpe             = (float*)alloc((RR + 1) * 4);
  unsigned short* seluEnc= (unsigned short*)alloc((size_t)BB * SS * HH * 2);
  float* encF            = (float*)alloc((size_t)BB * SS * 100 * 4);
  float* score           = (float*)alloc((size_t)BB * SS * 4);
  unsigned short* ctxbf  = (unsigned short*)alloc((size_t)BB * HH * 2);
  float* xctx            = (float*)alloc((size_t)BB * EE * 4);
  float* hbuf            = (float*)alloc((size_t)BB * HH * 4);
  unsigned short* hbf    = (unsigned short*)alloc((size_t)BB * HH * 2);
  unsigned short* shbf   = (unsigned short*)alloc((size_t)BB * HH * 2);
  unsigned short* embbf  = (unsigned short*)alloc((size_t)BB * EE * 2);
  unsigned short* xbf    = (unsigned short*)alloc((size_t)BB * EE * 2);
  float* gi              = (float*)alloc((size_t)BB * 3 * HH * 4);
  float* gh              = (float*)alloc((size_t)BB * 3 * HH * 4);
  float* pred            = (float*)alloc((size_t)BB * (RR + 1) * 4);
  float* outF            = (float*)alloc((size_t)BB * 100 * 4);
  float* eos             = (float*)alloc(BB * 4);
  int* act               = (int*)alloc(BB * 4);
  int* word              = (int*)alloc(BB * 4);
  (void)ws_size; (void)n_in; (void)in_sizes; (void)out_size;

  const int T = 256;
  auto blk = [](int n, int t) { return (n + t - 1) / t; };
  auto gemm = [&](const unsigned short* A, const unsigned short* B, float* C,
                  unsigned short* Cbf, const float* bias, const float* Cadd,
                  int M, int N, int K) {
    int waves = (M >> 4) * ((N + 63) >> 6);
    k_gemm_bf16<<<blk(waves, 8), 256, 0, stream>>>(
        (const __bf16*)A, (const __bf16*)B, C, Cbf, bias, Cadd, M, N, K);
  };

  // ---- one-time weight conversion / folding ----
  k_conv_bf16<<<blk(EE * EE, T), T, 0, stream>>>(W_comb, WcEmb, EE, EE, HH + EE, 0);
  k_conv_bf16<<<blk(EE * HH, T), T, 0, stream>>>(W_comb, Wcc, EE, HH, HH + EE, EE);
  k_conv_bf16<<<blk(NCELL * 3 * HH * EE, T), T, 0, stream>>>(W_ih, Wih, NCELL * 3 * HH, EE, EE, 0);
  k_conv_bf16<<<blk(NCELL * 3 * HH * HH, T), T, 0, stream>>>(W_hh, Whh, NCELL * 3 * HH, HH, HH, 0);
  k_conv_bf16<<<blk(100 * HH, T), T, 0, stream>>>(W_fuse, WfOut, 100, HH, 2 * HH, 0);
  k_conv_bf16<<<blk(100 * HH, T), T, 0, stream>>>(W_fuse, WfEnc, 100, HH, 2 * HH, HH);
  k_conv_bf16<<<blk(RR * HH, T), T, 0, stream>>>(W_pred, Wpe, RR, HH, HH, 0);
  k_conv_bf16<<<blk(HH, T), T, 0, stream>>>(W_eos, Wpe + (size_t)RR * HH, 1, HH, HH, 0);
  k_bpe<<<1, 64, 0, stream>>>(b_pred, b_eos, bpe);

  // ---- step-invariant precompute ----
  k_selu_bf16<<<blk(BB * SS * HH, T), T, 0, stream>>>(enc, seluEnc, BB * SS * HH);
  k_score<<<blk(BB * SS * 32, T), T, 0, stream>>>(enc, W_attn, score);
  k_ctx<<<BB, 256, 0, stream>>>(score, enc, ctxbf);
  gemm(ctxbf, Wcc, xctx, nullptr, b_comb, nullptr, BB, EE, HH);           // x_ctx (+b_comb)
  gemm(seluEnc, WfEnc, encF, nullptr, nullptr, nullptr, BB * SS, 100, HH); // encF (no bias)

  // ---- initial state ----
  k_zero<<<blk(BB * HH, T), T, 0, stream>>>(hbuf, BB * HH);
  k_sos<<<blk(BB * EE, T), T, 0, stream>>>(sos_emb, embbf);

  const size_t REL_OFF = 0;
  const size_t COPY_OFF = (size_t)NCELL * BB * (RR + 1);
  const size_t ACT_OFF = COPY_OFF + (size_t)2 * NCELL * BB * (SS + 1);

  for (int c = 0; c < NCELL; ++c) {
    k_cell_init<<<blk(BB * HH, T), T, 0, stream>>>(dec_state, hbuf, hbf);
    for (int t = 0; t < 3; ++t) {
      // x = emb@WcEmb^T + x_ctx  (bf16 out only)
      gemm(embbf, WcEmb, nullptr, xbf, nullptr, xctx, BB, EE, EE);
      // GRU gates
      gemm(xbf, Wih + (size_t)c * 3 * HH * EE, gi, nullptr, b_ih + (size_t)c * 3 * HH, nullptr, BB, 3 * HH, EE);
      gemm(hbf, Whh + (size_t)c * 3 * HH * HH, gh, nullptr, b_hh + (size_t)c * 3 * HH, nullptr, BB, 3 * HH, HH);
      k_gru<<<blk(BB * HH, T), T, 0, stream>>>(gi, gh, hbuf, hbf, shbf);

      float* act_slot = out + ACT_OFF + (size_t)(c * 3 + t) * BB;
      if (t == 0) {
        gemm(hbf, Wpe, pred, nullptr, bpe, nullptr, BB, RR + 1, HH);
        k_pred_fin<<<blk(BB * 32, T), T, 0, stream>>>(
            pred, out + REL_OFF + (size_t)c * BB * (RR + 1), act_slot, act);
        k_gather<<<BB, 128, 0, stream>>>(rel_emb, act, embbf);
      } else {
        k_eos<<<blk(BB * 32, T), T, 0, stream>>>(hbuf, W_eos, b_eos, eos);
        gemm(shbf, WfOut, outF, nullptr, b_fuse, nullptr, BB, 100, HH);
        k_copy_fin<<<BB, 128, 0, stream>>>(
            outF, encF, W_copy, b_copy, eos, sentence,
            out + COPY_OFF + (size_t)(c * 2 + (t - 1)) * BB * (SS + 1), act_slot, word);
        k_gather<<<BB, 128, 0, stream>>>(word_emb, word, embbf);
      }
    }
  }
}
